// se3ACN_80453327389084
// MI455X (gfx1250) — matrix operand order repaired and checked by reading, compiled, not run on hardware
//
#include <hip/hip_runtime.h>
#include <hip/hip_bf16.h>
#include <cstdint>
#include <cstddef>

typedef _Float16 f16;
typedef _Float16 v16h __attribute__((ext_vector_type(16)));
typedef float    v8f  __attribute__((ext_vector_type(8)));
typedef unsigned int u32x8 __attribute__((ext_vector_type(8)));

#define BB      16
#define NN      64
#define NPAIR   (BB*NN*NN)      // 65536
#define EMB     32
#define CDIM    24
#define NL      3
#define NBASIS  10
#define HH      150
#define HPAD    160
#define KD      (HPAD/2)        // 80 dwords of f16 per K row
#define BPITCH  85              // LDS row pitch in dwords: 85 mod 64 = 21, gcd(21,64)=1 -> conflict-free
#define LO      72              // NL*CDIM
#define LOPAD   80
#define ROUT    (NL*CDIM*EMB)   // 2304
#define CR      219
#define C2R     438
#define RADIUSF 2.0f

// softplus via hardware transcendentals (v_exp_f32 / v_log_f32)
__device__ __forceinline__ float sp5(float x) {            // softplus(beta=5)/5
    float bx = 5.0f * x;
    return (bx > 20.0f) ? x : __logf(1.0f + __expf(bx)) * 0.2f;
}
__device__ __forceinline__ float sp1(float x) {            // softplus
    return (x > 20.0f) ? x : __logf(1.0f + __expf(x));
}

// ---------------------------------------------------------------------------
// Repack a [150x150] f32 row-major (KxN) weight into [160x160] f16
// column-major (N-major, K contiguous) so B-fragment dwords pack (k,k+1).
// ---------------------------------------------------------------------------
__global__ void k_wconv(const float* __restrict__ W, f16* __restrict__ Wt) {
    int idx = blockIdx.x * blockDim.x + threadIdx.x;      // 160*160
    if (idx >= HPAD * HPAD) return;
    int n = idx / HPAD, k = idx - n * HPAD;
    f16 v = (f16)0.0f;
    if (k < HH && n < HH) v = (f16)W[k * HH + n];
    Wt[(size_t)n * HPAD + k] = v;
}

// ---------------------------------------------------------------------------
// Per-pair geometry: r, mask, Y(l=0..2, 9 comps), cos^2 basis, radial layer0.
// One thread per pair; rW0/rb0 staged in LDS.
// ---------------------------------------------------------------------------
__global__ void k_geom_l0(const float* __restrict__ xyz,
                          const float* __restrict__ rW0,
                          const float* __restrict__ rb0,
                          float* __restrict__ Y, float* __restrict__ maskv,
                          f16* __restrict__ h0) {
    __shared__ float sW0[NBASIS * HH];
    __shared__ float sb0[HH];
    int tid = threadIdx.x;
    for (int i = tid; i < NBASIS * HH; i += blockDim.x) sW0[i] = rW0[i];
    for (int i = tid; i < HH; i += blockDim.x) sb0[i] = rb0[i];
    __syncthreads();

    int p = blockIdx.x * blockDim.x + tid;                // 0..65535
    int b = p >> 12, a = (p >> 6) & 63, n = p & 63;
    const float* pa = xyz + ((size_t)b * NN + a) * 3;
    const float* pn = xyz + ((size_t)b * NN + n) * 3;
    float dx = pa[0] - pn[0], dy = pa[1] - pn[1], dz = pa[2] - pn[2];
    float r  = sqrtf(dx*dx + dy*dy + dz*dz + 1e-12f);
    float mk = (r < RADIUSF && r > 1e-6f) ? 1.0f : 0.0f;
    maskv[p] = mk;
    float inv = 1.0f / r;
    float x = dx * inv, y = dy * inv, z = dz * inv;

    const float c1 = 0.4886025119029199f;
    const float c2a = 1.0925484305920792f, c2b = 0.31539156525252005f, c2c = 0.5462742152960396f;
    float* Yp = Y + (size_t)p * 9;
    Yp[0] = 0.28209479177387814f;
    Yp[1] = c1 * y; Yp[2] = c1 * z; Yp[3] = c1 * x;
    Yp[4] = c2a * x * y; Yp[5] = c2a * y * z; Yp[6] = c2b * (3.0f*z*z - 1.0f);
    Yp[7] = c2a * x * z; Yp[8] = c2c * (x*x - y*y);

    // cosine basis
    float bas[NBASIS];
    const float step = RADIUSF / (NBASIS - 1);
    #pragma unroll
    for (int j = 0; j < NBASIS; ++j) {
        float d = (r - step * (float)j) / step;
        float ad = fabsf(d);
        float cc = __cosf(1.5707963267948966f * d);
        bas[j] = (ad < 1.0f) ? cc * cc : 0.0f;
    }
    // layer0 : 10 -> 150, softplus(beta=5)
    f16* hp = h0 + (size_t)p * HPAD;
    for (int k = 0; k < HH; ++k) {
        float acc = sb0[k];
        #pragma unroll
        for (int j = 0; j < NBASIS; ++j) acc += bas[j] * sW0[j * HH + k];
        hp[k] = (f16)sp5(acc);
    }
    #pragma unroll
    for (int k = HH; k < HPAD; ++k) hp[k] = (f16)0.0f;
}

// ---------------------------------------------------------------------------
// WMMA GEMM: [65536 x 160] f16 @ [160 x 160] f16(col-major) -> softplus5 -> f16.
// B matrix staged once per block in LDS (conflict-free pitch); each wave
// computes a 16x32 tile (2 accumulators per A fragment), K in 5 chunks of 32.
// ---------------------------------------------------------------------------
__global__ void k_gemm_sp(const f16* __restrict__ X, const f16* __restrict__ Wt,
                          const float* __restrict__ bias, f16* __restrict__ Out) {
    __shared__ unsigned int Bs[HPAD * BPITCH];            // 160*85*4 = 54.4 KB
    int tid = threadIdx.x;
    const unsigned int* Wd = (const unsigned int*)Wt;
    for (int i = tid; i < HPAD * KD; i += 256) {
        int n2 = i / KD, kd = i - n2 * KD;
        Bs[n2 * BPITCH + kd] = Wd[i];
    }
    __syncthreads();

    int wave = tid >> 5, lane = tid & 31;
    int gw   = blockIdx.x * 8 + wave;                     // 4096 mtiles * 5 ngroups
    int mtile = gw / 5, ng = gw - mtile * 5;
    int half = lane >> 4, l15 = lane & 15;
    const f16* xr = X + ((size_t)mtile * 16 + l15) * HPAD;
    const unsigned int* b0 = Bs + (ng * 32 + l15) * BPITCH;
    const unsigned int* b1 = b0 + 16 * BPITCH;

    v8f c0 = {}, c1 = {};
    #pragma unroll
    for (int kk = 0; kk < 5; ++kk) {
        u32x8 au, bu0, bu1;
        #pragma unroll
        for (int v = 0; v < 8; ++v) {
            int ka = kk * 32 + half * 16 + v * 2;
            au[v] = *(const unsigned int*)(xr + ka);
            int kd = kk * 16 + half * 8 + v;              // dword index of (k,k+1)
            bu0[v] = b0[kd];
            bu1[v] = b1[kd];
        }
        v16h a  = __builtin_bit_cast(v16h, au);
        v16h bm0 = __builtin_bit_cast(v16h, bu0);
        v16h bm1 = __builtin_bit_cast(v16h, bu1);
        c0 = __builtin_amdgcn_wmma_f32_16x16x32_f16(false, a, false, bm0, (short)0, c0, false, false);
        c1 = __builtin_amdgcn_wmma_f32_16x16x32_f16(false, a, false, bm1, (short)0, c1, false, false);
    }
    #pragma unroll
    for (int v = 0; v < 8; ++v) {
        int m  = v + 8 * half;
        size_t ro = (size_t)mtile * 16 + m;
        int co0 = ng * 32 + l15;
        int co1 = co0 + 16;
        f16 v0 = (f16)0.0f, v1 = (f16)0.0f;
        if (co0 < HH) v0 = (f16)sp5(c0[v] + bias[co0]);
        if (co1 < HH) v1 = (f16)sp5(c1[v] + bias[co1]);
        Out[ro * HPAD + co0] = v0;
        Out[ro * HPAD + co1] = v1;
    }
}

// ---------------------------------------------------------------------------
// Pre-contract rW3 with embeddings: W3f[bn][k, lo] = sum_i rW3[k,lo,i]*f[bn,i]
// stored f16, B-fragment layout (lo-major, k contiguous), padded 80x160.
// Also b3f[bn][lo] = sum_i rb3[lo,i]*f[bn,i].
// ---------------------------------------------------------------------------
__global__ void k_w3f(const int* __restrict__ Z, const float* __restrict__ embW,
                      const float* __restrict__ rW3, const float* __restrict__ rb3,
                      f16* __restrict__ W3f, float* __restrict__ b3f) {
    int bn = blockIdx.x;                                  // b*64+n
    int tid = threadIdx.x;
    __shared__ float fsh[EMB];
    if (tid < EMB) fsh[tid] = embW[(size_t)Z[bn] * EMB + tid];
    __syncthreads();
    if (tid < LO) {
        float acc = 0.0f;
        #pragma unroll
        for (int i = 0; i < EMB; ++i) acc += rb3[tid * EMB + i] * fsh[i];
        b3f[(size_t)bn * LO + tid] = acc;
    }
    f16* out = W3f + (size_t)bn * LOPAD * HPAD;
    for (int idx = tid; idx < LOPAD * HPAD; idx += blockDim.x) {
        int lo = idx / HPAD, k = idx - lo * HPAD;
        float acc = 0.0f;
        if (lo < LO && k < HH) {
            const float* wr = rW3 + (size_t)k * ROUT + lo * EMB;
            #pragma unroll
            for (int i = 0; i < EMB; ++i) acc += wr[i] * fsh[i];
        }
        out[(size_t)lo * HPAD + k] = (f16)acc;
    }
}

// ---------------------------------------------------------------------------
// Per-(b,n) WMMA GEMM: 64 rows (a=0..63, strided pairs) x [160 -> 80],
// B matrix (W3f[bn]) staged in LDS; fused bias + mask:
//   g[p, lo] = (h2[p] @ W3f[bn] + b3f[bn]) * mask[p]
// ---------------------------------------------------------------------------
__global__ void k_pair_gemm(const f16* __restrict__ X, const f16* __restrict__ W3f,
                            const float* __restrict__ b3f, const float* __restrict__ maskv,
                            float* __restrict__ g) {
    __shared__ unsigned int Bs[LOPAD * BPITCH];           // 80*85*4 = 27.2 KB
    int bn = blockIdx.x;
    int b = bn >> 6, n = bn & 63;
    int tid = threadIdx.x;
    const unsigned int* Wd = (const unsigned int*)(W3f + (size_t)bn * LOPAD * HPAD);
    for (int i = tid; i < LOPAD * KD; i += 256) {
        int n2 = i / KD, kd = i - n2 * KD;
        Bs[n2 * BPITCH + kd] = Wd[i];
    }
    __syncthreads();

    int wave = tid >> 5, lane = tid & 31;
    int half = lane >> 4, l15 = lane & 15;

    for (int t = wave; t < 20; t += 8) {                  // 4 M-tiles x 5 N-tiles
        int mt = t / 5, nt = t - mt * 5;
        int arow = mt * 16 + l15;
        size_t prow = (size_t)b * 4096 + (size_t)arow * 64 + n;
        const f16* xr = X + prow * HPAD;
        const unsigned int* br = Bs + (nt * 16 + l15) * BPITCH;
        v8f c = {};
        #pragma unroll
        for (int kk = 0; kk < 5; ++kk) {
            u32x8 au, bu;
            #pragma unroll
            for (int v = 0; v < 8; ++v) {
                int ka = kk * 32 + half * 16 + v * 2;
                au[v] = *(const unsigned int*)(xr + ka);
                bu[v] = br[kk * 16 + half * 8 + v];
            }
            v16h a    = __builtin_bit_cast(v16h, au);
            v16h bmat = __builtin_bit_cast(v16h, bu);
            c = __builtin_amdgcn_wmma_f32_16x16x32_f16(false, a, false, bmat,
                                                       (short)0, c, false, false);
        }
        #pragma unroll
        for (int v = 0; v < 8; ++v) {
            int m = v + 8 * half;
            size_t po = (size_t)b * 4096 + (size_t)(mt * 16 + m) * 64 + n;
            int lo = nt * 16 + l15;
            if (lo < LO)
                g[po * LO + lo] = (c[v] + b3f[(size_t)bn * LO + lo]) * maskv[po];
        }
    }
}

// ---------------------------------------------------------------------------
// Equivariant neighbor reduction: feats[atom, 3 + (l,o,m)] layout of reference
// ---------------------------------------------------------------------------
__global__ void k_conv(const float* __restrict__ g, const float* __restrict__ Y,
                       const float* __restrict__ body23, float* __restrict__ feats) {
    int atom = blockIdx.x;                                // b*64+a
    int c = threadIdx.x;
    if (c >= CR) return;
    if (c < 3) { feats[(size_t)atom * CR + c] = body23[(size_t)atom * 3 + c]; return; }
    int cc = c - 3;
    int lo, yidx;
    if (cc < 24)      { lo = cc;                       yidx = 0; }
    else if (cc < 96) { int t2 = cc - 24; lo = CDIM   + t2 / 3; yidx = 1 + t2 % 3; }
    else              { int t2 = cc - 96; lo = 2*CDIM + t2 / 5; yidx = 4 + t2 % 5; }
    float acc = 0.0f;
    for (int n = 0; n < NN; ++n) {
        size_t p = (size_t)atom * NN + n;
        acc += g[p * LO + lo] * Y[p * 9 + yidx];
    }
    feats[(size_t)atom * CR + c] = acc * 0.125f;          // 1/sqrt(64)
}

// ---------------------------------------------------------------------------
// AtomResiduals: out438 = concat(x, x + sp5(x @ resW + resb))
// ---------------------------------------------------------------------------
__global__ void k_resid(const float* __restrict__ feats, const float* __restrict__ resW,
                        const float* __restrict__ resb, float* __restrict__ out438) {
    int row = blockIdx.x;
    int j = threadIdx.x;
    __shared__ float xs[CR];
    if (j < CR) xs[j] = feats[(size_t)row * CR + j];
    __syncthreads();
    if (j >= CR) return;
    float t = resb[j];
    for (int k = 0; k < CR; ++k) t += xs[k] * resW[(size_t)k * CR + j];
    out438[(size_t)row * C2R + j]      = xs[j];
    out438[(size_t)row * C2R + CR + j] = xs[j] + sp5(t);
}

// L2 pool over atoms
__global__ void k_pool(const float* __restrict__ out438, float* __restrict__ pooled) {
    int idx = blockIdx.x * blockDim.x + threadIdx.x;      // 16*438
    if (idx >= BB * C2R) return;
    int b = idx / C2R, d = idx - b * C2R;
    float acc = 0.0f;
    for (int a = 0; a < NN; ++a) {
        float v = out438[((size_t)b * NN + a) * C2R + d];
        acc += v * v;
    }
    pooled[idx] = sqrtf(acc);
}

// collate linear + softplus
__global__ void k_collate(const float* __restrict__ pooled, const float* __restrict__ cW,
                          const float* __restrict__ cb, float* __restrict__ hh) {
    int idx = blockIdx.x * blockDim.x + threadIdx.x;      // 16*128
    if (idx >= BB * 128) return;
    int b = idx / 128, ch = idx - b * 128;
    float acc = cb[ch];
    for (int k = 0; k < C2R; ++k) acc += pooled[(size_t)b * C2R + k] * cW[(size_t)k * 128 + ch];
    hh[idx] = sp1(acc);
}

// BatchNorm (batch stats) + softplus + output linear + sigmoid
__global__ void k_final(const float* __restrict__ hh, const float* __restrict__ bng,
                        const float* __restrict__ bnb, const float* __restrict__ oW,
                        const float* __restrict__ ob, float* __restrict__ out) {
    __shared__ float sacc[BB * 128];
    int c = threadIdx.x;                                  // 128 threads
    float mu = 0.0f, m2 = 0.0f;
    for (int b = 0; b < BB; ++b) { float v = hh[b * 128 + c]; mu += v; m2 += v * v; }
    mu *= (1.0f / BB);
    float var = m2 * (1.0f / BB) - mu * mu;
    float inv = rsqrtf(var + 1e-5f);
    for (int b = 0; b < BB; ++b) {
        float v = hh[b * 128 + c];
        float hn = sp1((v - mu) * inv * bng[c] + bnb[c]);
        sacc[b * 128 + c] = hn * oW[c];
    }
    __syncthreads();
    if (c < BB) {
        float s = ob[0];
        for (int k = 0; k < 128; ++k) s += sacc[c * 128 + k];
        out[c] = 1.0f / (1.0f + __expf(-s));
    }
}

// ---------------------------------------------------------------------------
extern "C" void kernel_launch(void* const* d_in, const int* in_sizes, int n_in,
                              void* d_out, int out_size, void* d_ws, size_t ws_size,
                              hipStream_t stream) {
    (void)in_sizes; (void)n_in; (void)out_size; (void)ws_size;
    const float* xyz    = (const float*)d_in[0];
    const int*   Z      = (const int*)  d_in[1];
    const float* body23 = (const float*)d_in[2];
    const float* embW   = (const float*)d_in[3];
    const float* rW0    = (const float*)d_in[4];
    const float* rb0    = (const float*)d_in[5];
    const float* rW1    = (const float*)d_in[6];
    const float* rb1    = (const float*)d_in[7];
    const float* rW2    = (const float*)d_in[8];
    const float* rb2    = (const float*)d_in[9];
    const float* rW3    = (const float*)d_in[10];
    const float* rb3    = (const float*)d_in[11];
    const float* resW   = (const float*)d_in[12];
    const float* resb   = (const float*)d_in[13];
    const float* cW     = (const float*)d_in[14];
    const float* cb     = (const float*)d_in[15];
    const float* bng    = (const float*)d_in[16];
    const float* bnb    = (const float*)d_in[17];
    const float* oW     = (const float*)d_in[18];
    const float* ob     = (const float*)d_in[19];
    float* out = (float*)d_out;

    char* w = (char*)d_ws;
    size_t off = 0;
    auto carve = [&](size_t bytes) { char* p = w + off; off = (off + bytes + 255) & ~(size_t)255; return p; };
    float* Yb     = (float*)carve((size_t)NPAIR * 9 * 4);
    float* maskb  = (float*)carve((size_t)NPAIR * 4);
    f16*   hA     = (f16*)  carve((size_t)NPAIR * HPAD * 2);
    f16*   hB     = (f16*)  carve((size_t)NPAIR * HPAD * 2);
    f16*   Wt1    = (f16*)  carve((size_t)HPAD * HPAD * 2);
    f16*   Wt2    = (f16*)  carve((size_t)HPAD * HPAD * 2);
    f16*   W3f    = (f16*)  carve((size_t)BB * NN * LOPAD * HPAD * 2);
    float* b3f    = (float*)carve((size_t)BB * NN * LO * 4);
    float* gbuf   = (float*)carve((size_t)NPAIR * LO * 4);
    float* feats  = (float*)carve((size_t)BB * NN * CR * 4);
    float* out438 = (float*)carve((size_t)BB * NN * C2R * 4);
    float* pooled = (float*)carve((size_t)BB * C2R * 4);
    float* hhb    = (float*)carve((size_t)BB * 128 * 4);

    k_wconv<<<(HPAD*HPAD + 255) / 256, 256, 0, stream>>>(rW1, Wt1);
    k_wconv<<<(HPAD*HPAD + 255) / 256, 256, 0, stream>>>(rW2, Wt2);
    k_geom_l0<<<NPAIR / 256, 256, 0, stream>>>(xyz, rW0, rb0, Yb, maskb, hA);
    k_gemm_sp<<<(NPAIR / 16) * 5 / 8, 256, 0, stream>>>(hA, Wt1, rb1, hB);
    k_gemm_sp<<<(NPAIR / 16) * 5 / 8, 256, 0, stream>>>(hB, Wt2, rb2, hA);
    k_w3f<<<BB * NN, 256, 0, stream>>>(Z, embW, rW3, rb3, W3f, b3f);
    k_pair_gemm<<<BB * NN, 256, 0, stream>>>(hA, W3f, b3f, maskb, gbuf);
    k_conv<<<BB * NN, 224, 0, stream>>>(gbuf, Yb, body23, feats);
    k_resid<<<BB * NN, 224, 0, stream>>>(feats, resW, resb, out438);
    k_pool<<<(BB * C2R + 255) / 256, 256, 0, stream>>>(out438, pooled);
    k_collate<<<(BB * 128 + 255) / 256, 256, 0, stream>>>(pooled, cW, cb, hhb);
    k_final<<<1, 128, 0, stream>>>(hhb, bng, bnb, oW, ob, out);
}
